// MultiHeadLSTMCell_40192303956539
// MI455X (gfx1250) — matrix operand order repaired and checked by reading, not compiled, not run on hardware
//
#include <hip/hip_runtime.h>

// ---------------------------------------------------------------------------
// MultiHeadLSTMCell for MI455X (gfx1250, wave32, WMMA bf16)
//   B=16, S=1024, H=HD=1024, 8 heads x 128 (gating is elementwise over HD)
//
// Phase 1: gx[g] = x_g @ W_gx + b_gx          (bf16 WMMA GEMM, 4x 16384x1024x1024)
// Phase 2: persistent scan kernel, 64 blocks, device-wide barrier per step.
//          Block w owns 16 columns; its slice of all 4 W_?h (128KB bf16) is
//          preloaded into the 320KB WGP LDS once; per-step h broadcast uses
//          global_load_async_to_lds_b128 (ASYNCcnt) to skip the VGPR round-trip.
// Phase 3: y = ys @ W_post + b_post           (bf16 WMMA GEMM -> f32 d_out)
// ---------------------------------------------------------------------------

#define Bb   16
#define Ss   1024
#define HDn  1024
#define BS   (Bb * Ss)          // 16384 rows
#define YN   (BS * HDn)         // 16777216 elements of y

typedef __attribute__((ext_vector_type(16))) __bf16        v16bf;
typedef __attribute__((ext_vector_type(8)))  float         v8f;
typedef __attribute__((ext_vector_type(4)))  unsigned int  u32x4;
typedef __attribute__((ext_vector_type(4)))  float         f32x4;

union Frag16 { v16bf v; u32x4 q[2]; };

__device__ __forceinline__ unsigned short f2bf(float x) {
  union { float f; unsigned u; } v; v.f = x;
  unsigned r = v.u + 0x7FFFu + ((v.u >> 16) & 1u);   // round-to-nearest-even
  return (unsigned short)(r >> 16);
}
__device__ __forceinline__ unsigned pack2bf(float lo, float hi) {
  return (unsigned)f2bf(lo) | ((unsigned)f2bf(hi) << 16);
}
__device__ __forceinline__ float bf2f(unsigned short u) {
  union { unsigned u; float f; } v; v.u = ((unsigned)u) << 16;
  return v.f;
}
__device__ __forceinline__ float sigf(float x) { return 1.0f / (1.0f + expf(-x)); }

// 32-bit wave-relative LDS byte address of a generic pointer (for async-to-LDS)
__device__ __forceinline__ unsigned lds_off(const void* p) {
  return (unsigned)(unsigned long long)
         (__attribute__((address_space(3))) const char*)(const char*)p;
}
// Per-lane async copy of 16 bytes global -> LDS, tracked with ASYNCcnt.
__device__ __forceinline__ void async_ld_b128(unsigned lds_dst_byte,
                                              unsigned voffset_byte,
                                              unsigned long long saddr) {
  asm volatile("global_load_async_to_lds_b128 %0, %1, %2"
               :: "v"(lds_dst_byte), "v"(voffset_byte), "s"(saddr) : "memory");
}
__device__ __forceinline__ void wait_async0() {
  asm volatile("s_wait_asynccnt 0x0" ::: "memory");
}

// ---------------------------------------------------------------------------
// Generic bf16-compute GEMM: C[MxN] = A[MxK](f32) @ W[KxN](f32) + bias
// Block: 256 threads = 8 waves; tile 128(M) x 128(N), K-step 32.
// Wave w computes rows [w*16, w*16+16) x all 128 columns (8 wmma accumulators).
// ---------------------------------------------------------------------------
#define LDP 40   // padded LDS row (32 + 8 halfwords) to spread banks, keeps 16B align

__global__ __launch_bounds__(256) void gemm_bf16(
    const float* __restrict__ A, const float* __restrict__ W,
    const float* __restrict__ bias, void* __restrict__ Cout,
    int M, int N, int K, int out_bf16)
{
  __shared__ unsigned short Al[128 * LDP];
  __shared__ unsigned short Wl[128 * LDP];   // transposed: [n][k]

  const int tid  = threadIdx.x;
  const int wave = tid >> 5, lane = tid & 31;
  const int r = lane & 15, hi = lane >> 4;
  const int m0 = blockIdx.y * 128, n0 = blockIdx.x * 128;
  const int lm = tid & 127, kq = tid >> 7;   // loader coords: row 0..127, k-half 0..1

  v8f acc[8];
  #pragma unroll
  for (int i = 0; i < 8; ++i)
    #pragma unroll
    for (int v = 0; v < 8; ++v) acc[i][v] = 0.0f;

  for (int k0 = 0; k0 < K; k0 += 32) {
    __syncthreads();
    {
      // A: 16 contiguous f32 -> 4x b128 loads, pack -> 2x b128 LDS stores
      const f32x4* Ap = (const f32x4*)&A[(size_t)(m0 + lm) * K + k0 + kq * 16];
      f32x4 a0 = Ap[0], a1 = Ap[1], a2 = Ap[2], a3 = Ap[3];
      u32x4 p0, p1;
      p0.x = pack2bf(a0.x, a0.y); p0.y = pack2bf(a0.z, a0.w);
      p0.z = pack2bf(a1.x, a1.y); p0.w = pack2bf(a1.z, a1.w);
      p1.x = pack2bf(a2.x, a2.y); p1.y = pack2bf(a2.z, a2.w);
      p1.z = pack2bf(a3.x, a3.y); p1.w = pack2bf(a3.z, a3.w);
      *(u32x4*)&Al[lm * LDP + kq * 16]     = p0;
      *(u32x4*)&Al[lm * LDP + kq * 16 + 8] = p1;

      // W: gather 16 strided f32 (transpose), pack -> 2x b128 LDS stores
      const float* Wp = &W[(size_t)(k0 + kq * 16) * N + n0 + lm];
      float w[16];
      #pragma unroll
      for (int j = 0; j < 16; ++j) w[j] = Wp[(size_t)j * N];
      u32x4 q0, q1;
      q0.x = pack2bf(w[0], w[1]);   q0.y = pack2bf(w[2], w[3]);
      q0.z = pack2bf(w[4], w[5]);   q0.w = pack2bf(w[6], w[7]);
      q1.x = pack2bf(w[8], w[9]);   q1.y = pack2bf(w[10], w[11]);
      q1.z = pack2bf(w[12], w[13]); q1.w = pack2bf(w[14], w[15]);
      *(u32x4*)&Wl[lm * LDP + kq * 16]     = q0;
      *(u32x4*)&Wl[lm * LDP + kq * 16 + 8] = q1;
    }
    __syncthreads();

    Frag16 a;  // A 16x32 bf16: lane(r,hi) -> M=r, K = hi*16 .. hi*16+15 (contiguous)
    a.q[0] = *(const u32x4*)&Al[(wave * 16 + r) * LDP + hi * 16];
    a.q[1] = *(const u32x4*)&Al[(wave * 16 + r) * LDP + hi * 16 + 8];
    #pragma unroll
    for (int nt = 0; nt < 8; ++nt) {
      Frag16 b; // B 32x16 bf16: lane(r,hi) -> N=r, K = hi*16 .. hi*16+15 (contiguous)
      b.q[0] = *(const u32x4*)&Wl[(nt * 16 + r) * LDP + hi * 16];
      b.q[1] = *(const u32x4*)&Wl[(nt * 16 + r) * LDP + hi * 16 + 8];
      acc[nt] = __builtin_amdgcn_wmma_f32_16x16x32_bf16(
          false, a.v, false, b.v, (short)0, acc[nt], false, false);
    }
  }

  // Epilogue. C/D layout: M = v + 8*hi, N = r (per 16x16 tile).
  #pragma unroll
  for (int nt = 0; nt < 8; ++nt) {
    int n = n0 + nt * 16 + r;
    float bv = bias[n];
    #pragma unroll
    for (int v = 0; v < 8; ++v) {
      int m = m0 + wave * 16 + v + 8 * hi;
      float val = acc[nt][v] + bv;
      if (out_bf16) ((unsigned short*)Cout)[(size_t)m * N + n] = f2bf(val);
      else          ((float*)Cout)[(size_t)m * N + n] = val;
    }
  }
}

// ---------------------------------------------------------------------------
// Weight prep: WhT[g][n][k] (bf16) = W_?h[k][n] (f32); one-time 16MB transpose.
// ---------------------------------------------------------------------------
__global__ __launch_bounds__(256) void prep_whT(
    const float* __restrict__ wF, const float* __restrict__ wI,
    const float* __restrict__ wZ, const float* __restrict__ wO,
    unsigned short* __restrict__ WhT)
{
  unsigned id = blockIdx.x * 256 + threadIdx.x;        // 0 .. 4M-1
  unsigned g = id >> 20, rem = id & 1048575u;
  unsigned n = rem >> 10, k = rem & 1023u;
  const float* w = (g == 0) ? wF : (g == 1) ? wI : (g == 2) ? wZ : wO;
  WhT[(size_t)id] = f2bf(w[(size_t)k * HDn + n]);
}

// ---------------------------------------------------------------------------
// Persistent recurrent scan. 64 blocks x 128 threads (4 waves, one per gate).
// LDS (dynamic, 169216 B < 320KB WGP):
//   Wl : [4 gates][16 cols][1024 k] bf16, rows padded to 1032  (132096 B)
//   Hl : [16 b][1024 k] bf16, rows padded to 1032              ( 33024 B)
//   Gl : [4][16][16] f32 gate preacts                          (  4096 B)
// ---------------------------------------------------------------------------
#define WROW 1032
#define HL_OFF  132096
#define GL_OFF  165120
#define SCAN_LDS 169216

__device__ __forceinline__ void grid_barrier(unsigned int* counter, unsigned int target) {
  __syncthreads();
  if (threadIdx.x == 0) {
    __hip_atomic_fetch_add(counter, 1u, __ATOMIC_RELEASE, __HIP_MEMORY_SCOPE_AGENT);
    while (__hip_atomic_load(counter, __ATOMIC_ACQUIRE, __HIP_MEMORY_SCOPE_AGENT) < target)
      __builtin_amdgcn_s_sleep(1);
  }
  __syncthreads();
  __builtin_amdgcn_fence(__ATOMIC_ACQUIRE, "agent");   // make peers' h stores visible
}

__global__ __launch_bounds__(128) void lstm_scan(
    const unsigned short* __restrict__ gx,    // [4][16384][1024] bf16 preacts
    const unsigned short* __restrict__ WhT,   // [4][1024][1024] bf16 (n-major)
    const float* __restrict__ h0, const float* __restrict__ c0,
    const float* __restrict__ bF, const float* __restrict__ bI,
    const float* __restrict__ bZ, const float* __restrict__ bO,
    float* __restrict__ ys,                   // [16384][1024] f32 (h history)
    unsigned short* __restrict__ hbuf,        // [2][16][1024] bf16 (double buffer)
    unsigned int* __restrict__ counter,
    float* __restrict__ dout,                 // y | c | h
    int S)
{
  extern __shared__ char smem[];
  unsigned short* Wl = (unsigned short*)smem;
  unsigned short* Hl = (unsigned short*)(smem + HL_OFF);
  float*          Gl = (float*)(smem + GL_OFF);

  const int tid = threadIdx.x, wave = tid >> 5, lane = tid & 31;
  const int r = lane & 15, hi = lane >> 4;
  const int nb = blockIdx.x * 16;             // this block's 16 columns of HD
  const unsigned wl_base = lds_off(Wl);
  const unsigned hl_base = lds_off(Hl);

  // --- one-time: async-stage this block's 128KB weight slice into LDS ------
  for (int i = tid; i < 8192; i += 128) {     // 8192 chunks of 16B
    int row = i >> 7, c16 = (i & 127) * 16;   // row = g*16 + c, byte off in row
    int g = row >> 4, c = row & 15;
    unsigned dst  = wl_base + (unsigned)row * (WROW * 2) + (unsigned)c16;
    unsigned voff = (unsigned)(g * HDn + nb + c) * (HDn * 2) + (unsigned)c16;
    async_ld_b128(dst, voff, (unsigned long long)WhT);
  }
  // init h double-buffer[0] with own columns of h0 (bf16)
  for (int i = tid; i < 256; i += 128) {
    int b = i >> 4, c = i & 15;
    hbuf[b * HDn + nb + c] = f2bf(h0[b * HDn + nb + c]);
  }
  // cell state lives in wave-0 registers for the whole scan
  float creg[8];
  #pragma unroll
  for (int v = 0; v < 8; ++v) creg[v] = c0[(v + 8 * hi) * HDn + nb + r];
  const float* bh = (wave == 0) ? bF : (wave == 1) ? bI : (wave == 2) ? bZ : bO;
  const float bval = bh[nb + r];
  const unsigned short* gxw = gx + (size_t)wave * BS * HDn;

  wait_async0();                   // weight slice resident in LDS
  __threadfence();                 // flush everyone's h0 stores
  grid_barrier(counter, 64u);      // all blocks initialized

  for (int t = 0; t < S; ++t) {
    const int cur = t & 1, nxt = cur ^ 1;
    // async-stage full h (32KB bf16) into LDS with bank padding
    {
      unsigned long long hsrc = (unsigned long long)hbuf + (unsigned)cur * 32768u;
      for (int i = tid; i < 2048; i += 128) {   // 2048 chunks of 16B
        unsigned b = (unsigned)i >> 7, k16 = ((unsigned)i & 127u) * 16u;
        async_ld_b128(hl_base + b * (WROW * 2) + k16, (unsigned)i * 16u, hsrc);
      }
      wait_async0();
    }
    __syncthreads();

    // preact_g(16x16) = h(16x1024) @ Wh_g[:, nb:nb+16]  — 32 chained bf16 WMMAs
    v8f acc;
    #pragma unroll
    for (int v = 0; v < 8; ++v) acc[v] = 0.0f;
    const unsigned short* WlW = Wl + wave * 16 * WROW;
    #pragma unroll
    for (int kk = 0; kk < 32; ++kk) {
      Frag16 a, b;
      a.q[0] = *(const u32x4*)&Hl[r * WROW + kk * 32 + hi * 16];
      a.q[1] = *(const u32x4*)&Hl[r * WROW + kk * 32 + hi * 16 + 8];
      b.q[0] = *(const u32x4*)&WlW[r * WROW + kk * 32 + hi * 16];
      b.q[1] = *(const u32x4*)&WlW[r * WROW + kk * 32 + hi * 16 + 8];
      acc = __builtin_amdgcn_wmma_f32_16x16x32_bf16(
          false, a.v, false, b.v, (short)0, acc, false, false);
    }

    // add precomputed x-projection + bias; exchange gates through LDS
    #pragma unroll
    for (int v = 0; v < 8; ++v) {
      int b_ = v + 8 * hi;
      float pre = acc[v] + bf2f(gxw[((size_t)(b_ * Ss + t)) * HDn + nb + r]) + bval;
      Gl[wave * 256 + b_ * 16 + r] = pre;
    }
    __syncthreads();

    if (wave == 0) {   // gating + state update (wave-uniform branch, no WMMA inside)
      #pragma unroll
      for (int v = 0; v < 8; ++v) {
        int b_ = v + 8 * hi, idx = b_ * 16 + r;
        float f = sigf(Gl[idx]);
        float i = sigf(Gl[256 + idx]);
        float z = tanhf(Gl[512 + idx]);
        float o = sigf(Gl[768 + idx]);
        creg[v] = f * creg[v] + i * z;
        float hN = o * tanhf(creg[v]);
        ys[((size_t)(b_ * Ss + t)) * HDn + nb + r] = hN;
        hbuf[nxt * (Bb * HDn) + b_ * HDn + nb + r] = f2bf(hN);
        if (t == S - 1) {
          dout[YN + b_ * HDn + nb + r]            = creg[v];   // c_final
          dout[YN + Bb * HDn + b_ * HDn + nb + r] = hN;        // h_final
        }
      }
    }
    grid_barrier(counter, 64u * (unsigned)(t + 2));
  }
}

// ---------------------------------------------------------------------------
// Host launcher. Workspace (~200.1 MB):
//   [0)            gx   : 4*16384*1024 bf16 = 134217728 B
//   [134217728)    ys   : 16384*1024  f32   =  67108864 B
//   [201326592)    WhT  : 4*1024*1024 bf16  =   8388608 B
//   [209715200)    hbuf : 2*16*1024  bf16   =     65536 B
//   [209780736)    barrier counter
// ---------------------------------------------------------------------------
extern "C" void kernel_launch(void* const* d_in, const int* in_sizes, int n_in,
                              void* d_out, int out_size, void* d_ws, size_t ws_size,
                              hipStream_t stream) {
  const float* f_in = (const float*)d_in[0];
  const float* i_in = (const float*)d_in[1];
  const float* z_in = (const float*)d_in[2];
  const float* o_in = (const float*)d_in[3];
  const float* c0   = (const float*)d_in[4];
  const float* h0   = (const float*)d_in[5];
  const float* W_Fx = (const float*)d_in[6];   const float* b_Fx = (const float*)d_in[7];
  const float* W_Ix = (const float*)d_in[8];   const float* b_Ix = (const float*)d_in[9];
  const float* W_Zx = (const float*)d_in[10];  const float* b_Zx = (const float*)d_in[11];
  const float* W_Ox = (const float*)d_in[12];  const float* b_Ox = (const float*)d_in[13];
  const float* W_Fh = (const float*)d_in[14];  const float* b_Fh = (const float*)d_in[15];
  const float* W_Ih = (const float*)d_in[16];  const float* b_Ih = (const float*)d_in[17];
  const float* W_Zh = (const float*)d_in[18];  const float* b_Zh = (const float*)d_in[19];
  const float* W_Oh = (const float*)d_in[20];  const float* b_Oh = (const float*)d_in[21];
  const float* W_post = (const float*)d_in[22]; const float* b_post = (const float*)d_in[23];

  char* ws = (char*)d_ws;
  unsigned short* gx   = (unsigned short*)(ws);
  float*          ys   = (float*)(ws + 134217728ull);
  unsigned short* WhT  = (unsigned short*)(ws + 201326592ull);
  unsigned short* hbuf = (unsigned short*)(ws + 209715200ull);
  unsigned int*   cnt  = (unsigned int*)(ws + 209780736ull);

  (void)hipMemsetAsync(cnt, 0, 256, stream);           // reset barrier (ws is not re-poisoned)

  prep_whT<<<16384, 256, 0, stream>>>(W_Fh, W_Ih, W_Zh, W_Oh, WhT);

  const float* xs[4] = {f_in, i_in, z_in, o_in};
  const float* Wx[4] = {W_Fx, W_Ix, W_Zx, W_Ox};
  const float* bx[4] = {b_Fx, b_Ix, b_Zx, b_Ox};
  for (int g = 0; g < 4; ++g) {
    gemm_bf16<<<dim3(HDn / 128, BS / 128), 256, 0, stream>>>(
        xs[g], Wx[g], bx[g], (void*)(gx + (size_t)g * BS * HDn),
        BS, HDn, HDn, /*out_bf16=*/1);
  }

  (void)hipFuncSetAttribute(reinterpret_cast<const void*>(&lstm_scan),
                            hipFuncAttributeMaxDynamicSharedMemorySize, SCAN_LDS);
  lstm_scan<<<64, 128, SCAN_LDS, stream>>>(gx, WhT, h0, c0,
                                           b_Fh, b_Ih, b_Zh, b_Oh,
                                           ys, hbuf, cnt, (float*)d_out, Ss);

  gemm_bf16<<<dim3(HDn / 128, BS / 128), 256, 0, stream>>>(
      ys, W_post, b_post, d_out, BS, HDn, HDn, /*out_bf16=*/0);
}
